// GRU_Single_Dense_Single_Skip_Dropout_78915729097333
// MI455X (gfx1250) — compile-verified, matrix operand order checked by reading, not run on hardware
//
#include <hip/hip_runtime.h>

typedef __bf16 bf16;
typedef __attribute__((ext_vector_type(16))) __bf16 v16bf;
typedef __attribute__((ext_vector_type(8)))  float  v8f;

#define B_   4096
#define T_   20
#define NIN_ 360
#define H_   512
#define NOUT_ 360
#define NINP 384            // NIN padded to 32
#define KFEATP 896          // (H + NIN) padded to 32
#define NOUTP 384           // NOUT padded to 64 (block-tile multiple)

union FragBF {
    uint4 q[2];
    v16bf v;
};

// Load a 16x32 bf16 fragment. Wave32 layout: lanes 0-15 = rows 0-15 holding
// K 0..7 / 16..23; lanes 16-31 = rows 0-15 holding K 8..15 / 24..31.
// byteOff is hoisted per-lane; the k-step lands in the load immediate offset.
__device__ __forceinline__ v16bf load_frag(const bf16* __restrict__ base, unsigned byteOff) {
    const char* pc = (const char*)base + byteOff;
    FragBF f;
    f.q[0] = *(const uint4*)(pc);        // K +0..7
    f.q[1] = *(const uint4*)(pc + 32);   // K +16..23
    return f.v;
}

__device__ __forceinline__ v8f wmma_bf16(v16bf a, v16bf b, v8f c) {
    return __builtin_amdgcn_wmma_f32_16x16x32_bf16(
        false, a, false, b, (short)0, c, false, false);
}

__device__ __forceinline__ float sigmoidf_(float x) {
    return 1.0f / (1.0f + __expf(-x));
}
__device__ __forceinline__ float tanhf_(float x) {
    x = fminf(fmaxf(x, -15.0f), 15.0f);
    float e2 = __expf(2.0f * x);
    return (e2 - 1.0f) / (e2 + 1.0f);
}

// ---------------- conversion / setup kernels ----------------

__global__ void pad_convert_kernel(const float* __restrict__ in, bf16* __restrict__ out,
                                   int R, int C, int Rpad, int Cpad) {
    int idx = blockIdx.x * blockDim.x + threadIdx.x;
    if (idx >= Rpad * Cpad) return;
    int r = idx / Cpad, c = idx - r * Cpad;
    float v = (r < R && c < C) ? in[(size_t)r * C + c] : 0.0f;
    out[idx] = (bf16)v;
}

__global__ void init_h_kernel(const float* __restrict__ h0,
                              float* __restrict__ hf, bf16* __restrict__ hb, int n) {
    int idx = blockIdx.x * blockDim.x + threadIdx.x;
    if (idx >= n) return;
    float v = h0[idx];
    hf[idx] = v;
    hb[idx] = (bf16)v;
}

__global__ void build_feat_kernel(const bf16* __restrict__ hb,
                                  const float* __restrict__ x,
                                  bf16* __restrict__ feat) {
    int idx = blockIdx.x * blockDim.x + threadIdx.x;
    if (idx >= B_ * KFEATP) return;
    int b = idx / KFEATP, c = idx - b * KFEATP;
    bf16 v;
    if (c < H_)             v = hb[(size_t)b * H_ + c];
    else if (c < H_ + NIN_) v = (bf16)x[(size_t)b * (T_ * NIN_) + (size_t)(T_ - 1) * NIN_ + (c - H_)];
    else                    v = (bf16)0.0f;
    feat[idx] = v;
}

// ---------------- GRU step: fused [x_t|h] projections + gates ----------------
// Wave tile: 32 rows x 16 gate-cols (2 A frags reuse 3 gate B frags -> 6 WMMA
// per 10 b128 loads). Block 256 = 8 waves: 2 m-waves x 4 n-waves = 64x64 tile.
// launch_bounds(256, 4): cap VGPRs so 2 workgroups co-reside per WGP.
// grid: (B/64, H/64)
__global__ __launch_bounds__(256, 4) void gru_step_kernel(
    const bf16* __restrict__ xt,      // x_pad + t*NINP ; row stride T_*NINP
    const bf16* __restrict__ h_bf,    // [B, H] bf16 (read)
    const float* __restrict__ h_f32,  // [B, H] f32  (read)
    const bf16* __restrict__ w_ih,    // [3H, NINP]
    const bf16* __restrict__ w_hh,    // [3H, H]
    const float* __restrict__ b_ih,   // [3H]
    const float* __restrict__ b_hh,   // [3H]
    float* __restrict__ h_f32_out,
    bf16* __restrict__ h_bf_out)
{
    const int lane = threadIdx.x & 31;
    const int wave = threadIdx.x >> 5;
    const int m0 = blockIdx.x * 64 + (wave & 1) * 32;
    const int j0 = blockIdx.y * 64 + (wave >> 1) * 16;

    v8f ar0 = {}, az0 = {}, axn0 = {}, ahn0 = {};
    v8f ar1 = {}, az1 = {}, axn1 = {}, ahn1 = {};

    const unsigned r   = (unsigned)(lane & 15);
    const unsigned hi8 = (unsigned)((lane >> 4) << 3);

    // ---- x segment: K = 384 (12 chunks, pipeline depth 4) ----
    {
        const unsigned XS = (unsigned)(T_ * NINP);
        const unsigned a0 = ((unsigned)(m0 + r) * XS + hi8) * 2u;   // byte offsets
        const unsigned a1 = a0 + 16u * XS * 2u;
        const unsigned br = ((unsigned)(j0 + r) * NINP + hi8) * 2u;
        const unsigned bz = br + (unsigned)(H_ * NINP) * 2u;
        const unsigned bn = br + (unsigned)(2 * H_ * NINP) * 2u;
        #pragma unroll 4
        for (unsigned k = 0; k < NINP * 2u; k += 64u) {
            v16bf A0 = load_frag(xt, a0 + k);
            v16bf A1 = load_frag(xt, a1 + k);
            v16bf Br = load_frag(w_ih, br + k);
            v16bf Bz = load_frag(w_ih, bz + k);
            v16bf Bn = load_frag(w_ih, bn + k);
            ar0  = wmma_bf16(A0, Br, ar0);
            az0  = wmma_bf16(A0, Bz, az0);
            axn0 = wmma_bf16(A0, Bn, axn0);
            ar1  = wmma_bf16(A1, Br, ar1);
            az1  = wmma_bf16(A1, Bz, az1);
            axn1 = wmma_bf16(A1, Bn, axn1);
        }
    }

    // ---- h segment: K = 512 (16 chunks, pipeline depth 4) ----
    {
        const unsigned a0 = ((unsigned)(m0 + r) * H_ + hi8) * 2u;
        const unsigned a1 = a0 + 16u * H_ * 2u;
        const unsigned br = ((unsigned)(j0 + r) * H_ + hi8) * 2u;
        const unsigned bz = br + (unsigned)(H_ * H_) * 2u;
        const unsigned bn = br + (unsigned)(2 * H_ * H_) * 2u;
        #pragma unroll 4
        for (unsigned k = 0; k < H_ * 2u; k += 64u) {
            v16bf A0 = load_frag(h_bf, a0 + k);
            v16bf A1 = load_frag(h_bf, a1 + k);
            v16bf Br = load_frag(w_hh, br + k);
            v16bf Bz = load_frag(w_hh, bz + k);
            v16bf Bn = load_frag(w_hh, bn + k);
            ar0  = wmma_bf16(A0, Br, ar0);
            az0  = wmma_bf16(A0, Bz, az0);
            ahn0 = wmma_bf16(A0, Bn, ahn0);
            ar1  = wmma_bf16(A1, Br, ar1);
            az1  = wmma_bf16(A1, Bz, az1);
            ahn1 = wmma_bf16(A1, Bn, ahn1);
        }
    }

    // ---- epilogue: gates + state update ----
    const int j  = j0 + (lane & 15);
    const int hi = lane >> 4;
    const float bias_r  = b_ih[j] + b_hh[j];
    const float bias_z  = b_ih[H_ + j] + b_hh[H_ + j];
    const float bias_xn = b_ih[2 * H_ + j];
    const float bias_hn = b_hh[2 * H_ + j];

    #pragma unroll
    for (int sub = 0; sub < 2; ++sub) {
        const v8f& vr  = sub ? ar1  : ar0;
        const v8f& vz  = sub ? az1  : az0;
        const v8f& vxn = sub ? axn1 : axn0;
        const v8f& vhn = sub ? ahn1 : ahn0;
        #pragma unroll
        for (int e = 0; e < 8; ++e) {
            int m = m0 + sub * 16 + e + 8 * hi;     // C/D layout
            unsigned off = (unsigned)m * H_ + (unsigned)j;
            float hv = h_f32[off];
            float rg = sigmoidf_(vr[e] + bias_r);
            float zg = sigmoidf_(vz[e] + bias_z);
            float ng = tanhf_(vxn[e] + bias_xn + rg * (vhn[e] + bias_hn));
            float hnew = (1.0f - zg) * ng + zg * hv;
            h_f32_out[off] = hnew;
            h_bf_out [off] = (bf16)hnew;
        }
    }
}

// ---------------- dense1: relu(feat @ w1^T + b1) -> bf16 hid ----------------
// Wave tile 32x32 (4 WMMA per 8 b128). Block 8 waves: 2m x 4n = 64 x 128.
// grid: (B/64, H/128)
__global__ __launch_bounds__(256, 4) void dense1_kernel(
    const bf16* __restrict__ feat,   // [B, KFEATP]
    const bf16* __restrict__ w1,     // [H, KFEATP]
    const float* __restrict__ b1,    // [H]
    bf16* __restrict__ hid)          // [B, H]
{
    const int lane = threadIdx.x & 31;
    const int wave = threadIdx.x >> 5;
    const int m0 = blockIdx.x * 64 + (wave & 1) * 32;
    const int j0 = blockIdx.y * 128 + (wave >> 1) * 32;

    v8f c00 = {}, c01 = {}, c10 = {}, c11 = {};

    const unsigned r   = (unsigned)(lane & 15);
    const unsigned hi8 = (unsigned)((lane >> 4) << 3);
    const unsigned a0  = ((unsigned)(m0 + r) * KFEATP + hi8) * 2u;
    const unsigned a1  = a0 + 16u * KFEATP * 2u;
    const unsigned b0  = ((unsigned)(j0 + r) * KFEATP + hi8) * 2u;
    const unsigned b1o = b0 + 16u * KFEATP * 2u;

    #pragma unroll 4
    for (unsigned k = 0; k < KFEATP * 2u; k += 64u) {
        v16bf A0 = load_frag(feat, a0 + k);
        v16bf A1 = load_frag(feat, a1 + k);
        v16bf B0 = load_frag(w1, b0 + k);
        v16bf B1 = load_frag(w1, b1o + k);
        c00 = wmma_bf16(A0, B0, c00);
        c01 = wmma_bf16(A0, B1, c01);
        c10 = wmma_bf16(A1, B0, c10);
        c11 = wmma_bf16(A1, B1, c11);
    }

    const int j  = j0 + (lane & 15);
    const int hi = lane >> 4;
    const float bias0 = b1[j];
    const float bias1 = b1[j + 16];
    #pragma unroll
    for (int e = 0; e < 8; ++e) {
        int ma = m0 + e + 8 * hi;
        int mb = ma + 16;
        float v00 = c00[e] + bias0; v00 = v00 > 0.0f ? v00 : 0.0f;
        float v01 = c01[e] + bias1; v01 = v01 > 0.0f ? v01 : 0.0f;
        float v10 = c10[e] + bias0; v10 = v10 > 0.0f ? v10 : 0.0f;
        float v11 = c11[e] + bias1; v11 = v11 > 0.0f ? v11 : 0.0f;
        hid[(unsigned)ma * H_ + j]      = (bf16)v00;
        hid[(unsigned)ma * H_ + j + 16] = (bf16)v01;
        hid[(unsigned)mb * H_ + j]      = (bf16)v10;
        hid[(unsigned)mb * H_ + j + 16] = (bf16)v11;
    }
}

// ---------------- dense2: hid @ w2^T + b2 -> fp32 out [B, NOUT] ----------------
// Wave tile 32x16. Block 8 waves: 2m x 4n = 64 x 64. grid: (B/64, NOUTP/64)
__global__ __launch_bounds__(256, 4) void dense2_kernel(
    const bf16* __restrict__ hid,    // [B, H]
    const bf16* __restrict__ w2,     // [NOUTP, H] (rows >= NOUT zero)
    const float* __restrict__ b2,    // [NOUT]
    float* __restrict__ out)         // [B, NOUT]
{
    const int lane = threadIdx.x & 31;
    const int wave = threadIdx.x >> 5;
    const int m0 = blockIdx.x * 64 + (wave & 1) * 32;
    const int j0 = blockIdx.y * 64 + (wave >> 1) * 16;

    v8f c0 = {}, c1 = {};

    const unsigned r   = (unsigned)(lane & 15);
    const unsigned hi8 = (unsigned)((lane >> 4) << 3);
    const unsigned a0  = ((unsigned)(m0 + r) * H_ + hi8) * 2u;
    const unsigned a1  = a0 + 16u * H_ * 2u;
    const unsigned b0  = ((unsigned)(j0 + r) * H_ + hi8) * 2u;

    #pragma unroll 4
    for (unsigned k = 0; k < H_ * 2u; k += 64u) {
        v16bf A0 = load_frag(hid, a0 + k);
        v16bf A1 = load_frag(hid, a1 + k);
        v16bf B0 = load_frag(w2, b0 + k);
        c0 = wmma_bf16(A0, B0, c0);
        c1 = wmma_bf16(A1, B0, c1);
    }

    const int j  = j0 + (lane & 15);
    const int hi = lane >> 4;
    if (j < NOUT_) {
        const float bias = b2[j];
        #pragma unroll
        for (int e = 0; e < 8; ++e) {
            int ma = m0 + e + 8 * hi;
            out[(unsigned)ma * NOUT_ + j]        = c0[e] + bias;
            out[(unsigned)(ma + 16) * NOUT_ + j] = c1[e] + bias;
        }
    }
}

// ---------------- host launcher ----------------
extern "C" void kernel_launch(void* const* d_in, const int* in_sizes, int n_in,
                              void* d_out, int out_size, void* d_ws, size_t ws_size,
                              hipStream_t stream) {
    const float* x    = (const float*)d_in[0];   // [B,T,NIN]
    const float* h0   = (const float*)d_in[1];   // [1,B,H]
    const float* w_ih = (const float*)d_in[2];   // [3H,NIN]
    const float* w_hh = (const float*)d_in[3];   // [3H,H]
    const float* b_ih = (const float*)d_in[4];   // [3H]
    const float* b_hh = (const float*)d_in[5];   // [3H]
    const float* w1   = (const float*)d_in[6];   // [H, H+NIN]
    const float* b1   = (const float*)d_in[7];   // [H]
    const float* w2   = (const float*)d_in[8];   // [NOUT,H]
    const float* b2   = (const float*)d_in[9];   // [NOUT]
    float* out = (float*)d_out;

    char* p = (char*)d_ws;
    auto take = [&](size_t bytes) {
        char* r = p;
        p += (bytes + 255) & ~(size_t)255;
        return r;
    };
    bf16*  x_pad    = (bf16*) take((size_t)B_ * T_ * NINP * sizeof(bf16));
    bf16*  w_ih_pad = (bf16*) take((size_t)3 * H_ * NINP * sizeof(bf16));
    bf16*  w_hh_bf  = (bf16*) take((size_t)3 * H_ * H_ * sizeof(bf16));
    bf16*  w1_pad   = (bf16*) take((size_t)H_ * KFEATP * sizeof(bf16));
    bf16*  w2_pad   = (bf16*) take((size_t)NOUTP * H_ * sizeof(bf16));
    float* h_f32[2];
    h_f32[0] = (float*)take((size_t)B_ * H_ * sizeof(float));
    h_f32[1] = (float*)take((size_t)B_ * H_ * sizeof(float));
    bf16* h_bf[2];
    h_bf[0] = (bf16*)take((size_t)B_ * H_ * sizeof(bf16));
    h_bf[1] = (bf16*)take((size_t)B_ * H_ * sizeof(bf16));
    bf16* feat = (bf16*)take((size_t)B_ * KFEATP * sizeof(bf16));
    bf16* hid  = (bf16*)take((size_t)B_ * H_ * sizeof(bf16));

    const int TPB = 256;
    auto blocks = [](long long n, int t) { return (int)((n + t - 1) / t); };

    // 1) convert / pad inputs to bf16
    pad_convert_kernel<<<blocks((long long)B_ * T_ * NINP, TPB), TPB, 0, stream>>>(
        x, x_pad, B_ * T_, NIN_, B_ * T_, NINP);
    pad_convert_kernel<<<blocks((long long)3 * H_ * NINP, TPB), TPB, 0, stream>>>(
        w_ih, w_ih_pad, 3 * H_, NIN_, 3 * H_, NINP);
    pad_convert_kernel<<<blocks((long long)3 * H_ * H_, TPB), TPB, 0, stream>>>(
        w_hh, w_hh_bf, 3 * H_, H_, 3 * H_, H_);
    pad_convert_kernel<<<blocks((long long)H_ * KFEATP, TPB), TPB, 0, stream>>>(
        w1, w1_pad, H_, H_ + NIN_, H_, KFEATP);
    pad_convert_kernel<<<blocks((long long)NOUTP * H_, TPB), TPB, 0, stream>>>(
        w2, w2_pad, NOUT_, H_, NOUTP, H_);
    init_h_kernel<<<blocks((long long)B_ * H_, TPB), TPB, 0, stream>>>(
        h0, h_f32[0], h_bf[0], B_ * H_);

    // 2) recurrent steps
    int cur = 0;
    dim3 sgrid(B_ / 64, H_ / 64);
    for (int t = 0; t < T_; ++t) {
        gru_step_kernel<<<sgrid, TPB, 0, stream>>>(
            x_pad + (size_t)t * NINP,
            h_bf[cur], h_f32[cur],
            w_ih_pad, w_hh_bf, b_ih, b_hh,
            h_f32[1 - cur], h_bf[1 - cur]);
        cur ^= 1;
    }

    // 3) head
    build_feat_kernel<<<blocks((long long)B_ * KFEATP, TPB), TPB, 0, stream>>>(
        h_bf[cur], x, feat);
    dense1_kernel<<<dim3(B_ / 64, H_ / 128), TPB, 0, stream>>>(feat, w1_pad, b1, hid);
    dense2_kernel<<<dim3(B_ / 64, NOUTP / 64), TPB, 0, stream>>>(hid, w2_pad, b2, out);
}
